// SlotAttention_18141941858696
// MI455X (gfx1250) — matrix operand order repaired and checked by reading, not compiled, not run on hardware
//
#include <hip/hip_runtime.h>

// Slot Attention for MI455X (gfx1250), wave32, WMMA f16 16x16x32.
// Memory-bound workload (~77 GFLOP vs ~1.3 GB traffic @ 23.3 TB/s):
//  - K stored row-major f16 [n][d]  (A-fragment pairs along d contiguous)
//  - V stored transposed f16 [d][n] (B-fragment pairs along n contiguous)
//  - updates = (attn^T V) / colsum  -> single pass, no normalized-attn rewrite

#define BB 64
#define NN 4096
#define DD 256
#define SS 8
#define CH 32     // N chunks of 128 rows for the attention kernel
#define ITERS 3

typedef _Float16 h16;
typedef __attribute__((ext_vector_type(16))) _Float16 v16h;
typedef __attribute__((ext_vector_type(8)))  float    v8f;
typedef __attribute__((ext_vector_type(4)))  unsigned u32x4;

union Frag {
    u32x4 q[2];
    v16h  h;
};

__device__ __forceinline__ v8f wmma16(const Frag& a, const Frag& b, v8f c) {
    return __builtin_amdgcn_wmma_f32_16x16x32_f16(
        false, a.h, false, b.h, (short)0, c, false, false);
}

// ---------------------------------------------------------------- init slots
__global__ void k_init_slots(const float* __restrict__ noise,
                             const float* __restrict__ mu,
                             const float* __restrict__ lsig,
                             float* __restrict__ slots) {
    int i = blockIdx.x * 256 + threadIdx.x;   // B*S*D = 131072
    int d = i & (DD - 1);
    slots[i] = mu[d] + __expf(lsig[d]) * noise[i];
}

// ------------------------------------------- transpose+convert Wk, Wv to f16
__global__ void k_wT(const float* __restrict__ Wk, const float* __restrict__ Wv,
                     h16* __restrict__ wkT, h16* __restrict__ wvT) {
    int i = blockIdx.x * 256 + threadIdx.x;   // 65536 = 256*256
    int n = i >> 8, k = i & 255;              // out layout [n][k]
    wkT[i] = (h16)Wk[k * DD + n];
    wvT[i] = (h16)Wv[k * DD + n];
}

// ---------------------------- LN(inputs) -> f16, then K = xWk, V^T = (xWv)^T
__global__ __launch_bounds__(128)
void k_lnkv(const float* __restrict__ x,
            const float* __restrict__ g, const float* __restrict__ bta,
            const h16* __restrict__ wkT, const h16* __restrict__ wvT,
            h16* __restrict__ k16, h16* __restrict__ v16T) {
    __shared__ __align__(16) h16 x16[64 * DD];   // 32 KB normalized tile
    int b = blockIdx.y, rb = blockIdx.x * 64;
    int tid = threadIdx.x;
    int row = tid >> 1, hf = tid & 1;            // 2 threads per row

    const float* src = x + ((size_t)(b * NN + rb + row)) * DD + hf * 128;
    float s = 0.f, sq = 0.f;
    #pragma unroll
    for (int i = 0; i < 128; i += 4) {
        float4 v4 = *(const float4*)(src + i);
        s  += v4.x + v4.y + v4.z + v4.w;
        sq += v4.x * v4.x + v4.y * v4.y + v4.z * v4.z + v4.w * v4.w;
    }
    s  += __shfl_xor(s, 1, 32);
    sq += __shfl_xor(sq, 1, 32);
    float mean = s * (1.f / DD);
    float var  = sq * (1.f / DD) - mean * mean;
    float rstd = rsqrtf(var + 1e-5f);
    for (int i = 0; i < 128; ++i) {
        int d = hf * 128 + i;
        x16[row * DD + d] = (h16)((src[i] - mean) * rstd * g[d] + bta[d]);
    }
    __syncthreads();

    int w = tid >> 5, lane = tid & 31;
    int m   = lane & 15;
    int ahi = (lane < 16) ? 0 : 8;    // A fragment K-offset per lane half
    int bhi = (lane < 16) ? 0 : 8;    // B fragment K-offset per lane half

    for (int mat = 0; mat < 2; ++mat) {
        const h16* wT = mat ? wvT : wkT;
        v8f acc[16];
        #pragma unroll
        for (int ct = 0; ct < 16; ++ct) acc[ct] = (v8f){0,0,0,0,0,0,0,0};

        for (int kk = 0; kk < 8; ++kk) {
            Frag a;
            const h16* ap = &x16[(w * 16 + m) * DD + kk * 32 + ahi];
            a.q[0] = *(const u32x4*)ap;
            a.q[1] = *(const u32x4*)(ap + 16);
            #pragma unroll
            for (int ct = 0; ct < 16; ++ct) {
                Frag bf;
                const h16* bp = wT + (size_t)(ct * 16 + m) * DD + kk * 32 + bhi;
                bf.q[0] = *(const u32x4*)bp;
                bf.q[1] = *(const u32x4*)(bp + 16);
                acc[ct] = wmma16(a, bf, acc[ct]);
            }
        }
        #pragma unroll
        for (int ct = 0; ct < 16; ++ct) {
            int d = ct * 16 + m;
            #pragma unroll
            for (int r = 0; r < 8; ++r) {
                int rowg = rb + w * 16 + ((lane < 16) ? r : 8 + r);
                float vv = acc[ct][r];
                if (mat == 0)
                    k16[((size_t)b * NN + rowg) * DD + d] = (h16)vv;
                else
                    v16T[((size_t)b * DD + d) * NN + rowg] = (h16)vv;
            }
        }
    }
}

// ------------------------------------------- q = LN(slots) @ Wq, f16, padded
__global__ __launch_bounds__(256)
void k_q(const float* __restrict__ slots,
         const float* __restrict__ g, const float* __restrict__ bta,
         const float* __restrict__ Wq, h16* __restrict__ q16) {
    __shared__ float ss[SS * DD];
    __shared__ float sm[SS], sr[SS];
    int b = blockIdx.x, tid = threadIdx.x;
    #pragma unroll
    for (int s = 0; s < SS; ++s) ss[s * DD + tid] = slots[(b * SS + s) * DD + tid];
    __syncthreads();
    if (tid < SS) {
        float m = 0.f, q = 0.f;
        for (int i = 0; i < DD; ++i) { float v = ss[tid * DD + i]; m += v; q += v * v; }
        m *= (1.f / DD); q = q * (1.f / DD) - m * m;
        sm[tid] = m; sr[tid] = rsqrtf(q + 1e-5f);
    }
    __syncthreads();
    #pragma unroll
    for (int s = 0; s < SS; ++s)
        ss[s * DD + tid] = (ss[s * DD + tid] - sm[s]) * sr[s] * g[tid] + bta[tid];
    __syncthreads();
    #pragma unroll
    for (int s = 0; s < SS; ++s) {
        float acc = 0.f;
        for (int k = 0; k < DD; ++k) acc += ss[s * DD + k] * Wq[k * DD + tid];
        q16[((size_t)b * 16 + s) * DD + tid] = (h16)acc;
    }
    #pragma unroll
    for (int s = SS; s < 16; ++s)          // zero-pad slot rows 8..15
        q16[((size_t)b * 16 + s) * DD + tid] = (h16)0.f;
}

// ------------- logits->softmax->colsum + U_partial = attn^T V  (per chunk)
__global__ __launch_bounds__(256)
void k_attn(const h16* __restrict__ k16, const h16* __restrict__ q16,
            const h16* __restrict__ v16T,
            float* __restrict__ Upart, float* __restrict__ csPart,
            float* __restrict__ outAttn, int writeAttn) {
    __shared__ __align__(16) h16 alds[16 * 128];  // attn^T tile, f16
    __shared__ float scol[8];
    int chunk = blockIdx.x, b = blockIdx.y;
    int tid = threadIdx.x, w = tid >> 5, lane = tid & 31;
    int m   = lane & 15;
    int ahi = (lane < 16) ? 0 : 8;
    int bhi = (lane < 16) ? 0 : 8;
    int base = chunk * 128;

    if (tid < 8) scol[tid] = 0.f;
    __syncthreads();

    // logits tile [16 rows x 16 slots] per wave
    v8f acc = (v8f){0,0,0,0,0,0,0,0};
    const h16* krow = k16 + ((size_t)b * NN + base + w * 16 + m) * DD;
    const h16* qrow = q16 + ((size_t)b * 16 + m) * DD;
    for (int kk = 0; kk < 8; ++kk) {
        Frag a;
        const h16* ap = krow + kk * 32 + ahi;
        a.q[0] = *(const u32x4*)ap;
        a.q[1] = *(const u32x4*)(ap + 16);
        Frag bf;
        const h16* bp = qrow + kk * 32 + bhi;
        bf.q[0] = *(const u32x4*)bp;
        bf.q[1] = *(const u32x4*)(bp + 16);
        acc = wmma16(a, bf, acc);
    }

    // softmax over 8 slots (lanes 0..7 / 16..23 hold valid slots)
    const float scale = 0.0625f;   // 1/sqrt(256)
    int slot = m;
    float attnv[8];
    #pragma unroll
    for (int r = 0; r < 8; ++r) {
        float xv = acc[r] * scale;
        if (slot >= 8) xv = -1e30f;
        float mx = xv;
        mx = fmaxf(mx, __shfl_xor(mx, 1, 8));
        mx = fmaxf(mx, __shfl_xor(mx, 2, 8));
        mx = fmaxf(mx, __shfl_xor(mx, 4, 8));
        float e = __expf(xv - mx);
        float sum = e;
        sum += __shfl_xor(sum, 1, 8);
        sum += __shfl_xor(sum, 2, 8);
        sum += __shfl_xor(sum, 4, 8);
        attnv[r] = e / sum;
    }

    // per-wave column sums -> shared -> per-chunk partial
    float c = 0.f;
    #pragma unroll
    for (int r = 0; r < 8; ++r) c += attnv[r];
    c += __shfl_xor(c, 16, 32);
    if (lane < 8) atomicAdd(&scol[slot], c);

    // stage attn^T into LDS f16 (zeros for padded slots); last iter -> d_out
    int rofs = (lane < 16) ? 0 : 8;
    #pragma unroll
    for (int r = 0; r < 8; ++r) {
        int nl = w * 16 + r + rofs;
        alds[slot * 128 + nl] = (slot < 8) ? (h16)attnv[r] : (h16)0.f;
        if (writeAttn && slot < 8)
            outAttn[((size_t)b * SS + slot) * NN + base + nl] = attnv[r];
    }
    __syncthreads();

    // U_partial[16 x 256] = attn^T [16 x 128] @ V [128 x 256]; 2 col-tiles/wave
    #pragma unroll
    for (int cti = 0; cti < 2; ++cti) {
        int ctg = w * 2 + cti;
        v8f uacc = (v8f){0,0,0,0,0,0,0,0};
        #pragma unroll
        for (int ks = 0; ks < 4; ++ks) {
            Frag a;
            const h16* ap = &alds[m * 128 + ks * 32 + ahi];
            a.q[0] = *(const u32x4*)ap;
            a.q[1] = *(const u32x4*)(ap + 16);
            Frag bf;
            const h16* bp = v16T + ((size_t)b * DD + ctg * 16 + m) * NN
                                 + base + ks * 32 + bhi;
            bf.q[0] = *(const u32x4*)bp;
            bf.q[1] = *(const u32x4*)(bp + 16);
            uacc = wmma16(a, bf, uacc);
        }
        if (lane < 16) {
            #pragma unroll
            for (int r = 0; r < 8; ++r)
                Upart[(((size_t)b * CH + chunk) * SS + r) * DD + ctg * 16 + lane]
                    = uacc[r];
        }
    }
    if (tid < 8)
        csPart[((size_t)b * CH + chunk) * SS + tid] = scol[tid];
}

// --------------- reduce partials, GRU, LN, residual MLP (tiny, VALU, 64 WGs)
__global__ __launch_bounds__(256)
void k_update(float* __restrict__ slots,
              const float* __restrict__ Upart, const float* __restrict__ csPart,
              const float* __restrict__ w_ih, const float* __restrict__ w_hh,
              const float* __restrict__ b_ih, const float* __restrict__ b_hh,
              const float* __restrict__ lng, const float* __restrict__ lnb,
              const float* __restrict__ w1, const float* __restrict__ b1,
              const float* __restrict__ w2, const float* __restrict__ b2,
              float* __restrict__ outSlots, int writeOut) {
    __shared__ float ssl[SS * DD];       // slots (prev, then new)
    __shared__ float sup[SS * DD];       // updates, then LN(slots)
    __shared__ float sh[SS * 4 * DD];    // mlp hidden 8x1024
    __shared__ float scs[SS], smean[SS], srstd[SS];
    int b = blockIdx.x, tid = threadIdx.x;

    #pragma unroll
    for (int s = 0; s < SS; ++s) ssl[s * DD + tid] = slots[(b * SS + s) * DD + tid];
    if (tid < SS) {
        float c = 0.f;
        for (int ch = 0; ch < CH; ++ch) c += csPart[((size_t)b * CH + ch) * SS + tid];
        scs[tid] = c + 1e-8f;
    }
    __syncthreads();
    #pragma unroll
    for (int s = 0; s < SS; ++s) {
        float u = 0.f;
        for (int ch = 0; ch < CH; ++ch)
            u += Upart[(((size_t)b * CH + ch) * SS + s) * DD + tid];
        sup[s * DD + tid] = u / scs[s];
    }
    __syncthreads();

    // GRUCell (PyTorch semantics)
    float newsl[SS];
    #pragma unroll
    for (int s = 0; s < SS; ++s) {
        const float* up = &sup[s * DD];
        const float* hp = &ssl[s * DD];
        float gir = b_ih[tid], giz = b_ih[DD + tid], gin = b_ih[2 * DD + tid];
        float ghr = b_hh[tid], ghz = b_hh[DD + tid], ghn = b_hh[2 * DD + tid];
        for (int k = 0; k < DD; ++k) {
            float uv = up[k], hv = hp[k];
            gir += w_ih[(0 * DD + tid) * DD + k] * uv;
            giz += w_ih[(1 * DD + tid) * DD + k] * uv;
            gin += w_ih[(2 * DD + tid) * DD + k] * uv;
            ghr += w_hh[(0 * DD + tid) * DD + k] * hv;
            ghz += w_hh[(1 * DD + tid) * DD + k] * hv;
            ghn += w_hh[(2 * DD + tid) * DD + k] * hv;
        }
        float r = 1.f / (1.f + __expf(-(gir + ghr)));
        float z = 1.f / (1.f + __expf(-(giz + ghz)));
        float n = tanhf(gin + r * ghn);
        newsl[s] = (1.f - z) * n + z * hp[tid];
    }
    __syncthreads();
    #pragma unroll
    for (int s = 0; s < SS; ++s) ssl[s * DD + tid] = newsl[s];
    __syncthreads();

    if (tid < SS) {
        float m = 0.f, q = 0.f;
        for (int i = 0; i < DD; ++i) { float v = ssl[tid * DD + i]; m += v; q += v * v; }
        m *= (1.f / DD); q = q * (1.f / DD) - m * m;
        smean[tid] = m; srstd[tid] = rsqrtf(q + 1e-5f);
    }
    __syncthreads();
    #pragma unroll
    for (int s = 0; s < SS; ++s)
        sup[s * DD + tid] = (ssl[s * DD + tid] - smean[s]) * srstd[s] * lng[tid] + lnb[tid];
    __syncthreads();

    #pragma unroll
    for (int jj4 = 0; jj4 < 4; ++jj4) {
        int jj = jj4 * DD + tid;
        float bb1 = b1[jj];
        #pragma unroll
        for (int s = 0; s < SS; ++s) {
            float a = bb1;
            const float* xp = &sup[s * DD];
            for (int k = 0; k < DD; ++k) a += xp[k] * w1[k * 4 * DD + jj];
            sh[s * 4 * DD + jj] = fmaxf(a, 0.f);
        }
    }
    __syncthreads();
    #pragma unroll
    for (int s = 0; s < SS; ++s) {
        float a = b2[tid];
        const float* hp2 = &sh[s * 4 * DD];
        for (int k = 0; k < 4 * DD; ++k) a += hp2[k] * w2[k * DD + tid];
        float out = ssl[s * DD + tid] + a;
        slots[(b * SS + s) * DD + tid] = out;
        if (writeOut) outSlots[(b * SS + s) * DD + tid] = out;
    }
}

// ---------------------------------------------------------------------------
extern "C" void kernel_launch(void* const* d_in, const int* in_sizes, int n_in,
                              void* d_out, int out_size, void* d_ws, size_t ws_size,
                              hipStream_t stream) {
    (void)in_sizes; (void)n_in; (void)out_size; (void)ws_size;
    const float* inputs = (const float*)d_in[0];
    const float* noise  = (const float*)d_in[1];
    const float* mu     = (const float*)d_in[2];
    const float* lsig   = (const float*)d_in[3];
    const float* lin_g  = (const float*)d_in[4];
    const float* lin_b  = (const float*)d_in[5];
    const float* lns_g  = (const float*)d_in[6];
    const float* lns_b  = (const float*)d_in[7];
    const float* lnm_g  = (const float*)d_in[8];
    const float* lnm_b  = (const float*)d_in[9];
    const float* Wq     = (const float*)d_in[10];
    const float* Wk     = (const float*)d_in[11];
    const float* Wv     = (const float*)d_in[12];
    const float* w_ih   = (const float*)d_in[13];
    const float* w_hh   = (const float*)d_in[14];
    const float* b_ih   = (const float*)d_in[15];
    const float* b_hh   = (const float*)d_in[16];
    const float* w1     = (const float*)d_in[17];
    const float* b1     = (const float*)d_in[18];
    const float* w2     = (const float*)d_in[19];
    const float* b2     = (const float*)d_in[20];

    char* ws = (char*)d_ws;
    size_t off = 0;
    h16*   k16   = (h16*)(ws + off);   off += (size_t)BB * NN * DD * 2;
    h16*   v16T  = (h16*)(ws + off);   off += (size_t)BB * NN * DD * 2;
    h16*   wkT   = (h16*)(ws + off);   off += (size_t)DD * DD * 2;
    h16*   wvT   = (h16*)(ws + off);   off += (size_t)DD * DD * 2;
    h16*   q16   = (h16*)(ws + off);   off += (size_t)BB * 16 * DD * 2;
    float* slots = (float*)(ws + off); off += (size_t)BB * SS * DD * 4;
    float* Upart = (float*)(ws + off); off += (size_t)BB * CH * SS * DD * 4;
    float* csPart= (float*)(ws + off); off += (size_t)BB * CH * SS * 4;

    float* outSlots = (float*)d_out;
    float* outAttn  = outSlots + (size_t)BB * SS * DD;

    k_init_slots<<<BB * SS * DD / 256, 256, 0, stream>>>(noise, mu, lsig, slots);
    k_wT<<<DD * DD / 256, 256, 0, stream>>>(Wk, Wv, wkT, wvT);
    k_lnkv<<<dim3(NN / 64, BB), 128, 0, stream>>>(inputs, lin_g, lin_b,
                                                  wkT, wvT, k16, v16T);
    for (int it = 0; it < ITERS; ++it) {
        k_q<<<BB, 256, 0, stream>>>(slots, lns_g, lns_b, Wq, q16);
        k_attn<<<dim3(CH, BB), 256, 0, stream>>>(k16, q16, v16T, Upart, csPart,
                                                 outAttn, it == ITERS - 1);
        k_update<<<BB, 256, 0, stream>>>(slots, Upart, csPart, w_ih, w_hh,
                                         b_ih, b_hh, lnm_g, lnm_b,
                                         w1, b1, w2, b2,
                                         outSlots, it == ITERS - 1);
    }
}
